// Attention_7258494730720
// MI455X (gfx1250) — matrix operand order repaired and checked by reading, not compiled, not run on hardware
//
#include <hip/hip_runtime.h>
#include <hip/hip_bf16.h>

typedef __attribute__((ext_vector_type(16))) _Float16 v16h;
typedef __attribute__((ext_vector_type(8)))  _Float16 v8h;
typedef __attribute__((ext_vector_type(4)))  _Float16 v4h;
typedef __attribute__((ext_vector_type(8)))  float    v8f;
typedef __attribute__((ext_vector_type(4)))  float    v4f;
typedef __attribute__((ext_vector_type(4)))  int      v4i;

#define WMMA_F16(A, B, C) \
  __builtin_amdgcn_wmma_f32_16x16x32_f16(false, (A), false, (B), (short)0, (C), false, false)

// ---- async-to-LDS support (guarded; falls back to synchronous staging) ----
#if defined(__gfx1250__) && __has_builtin(__builtin_amdgcn_global_load_async_to_lds_b128)
#define HAS_ASYNC 1
#else
#define HAS_ASYNC 0
#endif

#if HAS_ASYNC
// Builtin prototype (from compiler diagnostic): params are v4i* in AS1 / AS3.
typedef __attribute__((address_space(1))) v4i* as1_v4i_p;
typedef __attribute__((address_space(3))) v4i* as3_v4i_p;
#if __has_builtin(__builtin_amdgcn_s_wait_asynccnt)
#define WAIT_ASYNC(n) __builtin_amdgcn_s_wait_asynccnt(n)
#else
#define WAIT_ASYNC(n) asm volatile("s_wait_asynccnt %0" ::"n"(n) : "memory")
#endif
#define ASYNC_B128(gp, lp) \
  __builtin_amdgcn_global_load_async_to_lds_b128((as1_v4i_p)(gp), (as3_v4i_p)(lp), 0, 0)
#else
#define WAIT_ASYNC(n)
#endif

static __device__ __forceinline__ v16h join16(v8h lo, v8h hi) {
  union { v16h v; v8h h[2]; } u;
  u.h[0] = lo; u.h[1] = hi;
  return u.v;
}

// Problem constants
static constexpr int Bsz = 4, L = 2048, D = 1024, NH = 16, HD = 64;
static constexpr int M = Bsz * L;        // 8192 rows
static constexpr int NQKV = 3 * NH * HD; // 3072 qkv output cols

// ---------------------------------------------------------------------------
// Kernel 1: QKV projection.  out(e,b,l,h) = X(8192x1024) x Wcat(1024x3072)+b
// 128x128 tile per 256-thread block; K-step 32; f16 WMMA, f32 accum.
// qkv ws layout: [e][m][h] f16  (e = 0..47, m = b*L+l, h = 0..63)
// (f32 sources need conversion during staging, so no raw async DMA here.)
// ---------------------------------------------------------------------------
__global__ __launch_bounds__(256) void qkv_gemm(const float* __restrict__ X,
                                                const float* __restrict__ W,
                                                const float* __restrict__ bias,
                                                _Float16* __restrict__ qkv) {
  __shared__ __align__(16) _Float16 sA[128 * 32];
  __shared__ __align__(16) _Float16 sB[128 * 32];
  const int t = threadIdx.x;
  const int wave = t >> 5, lane = t & 31, lh = lane & 15, hi = lane >> 4;
  const int m0 = blockIdx.x * 128;
  const int n0 = blockIdx.y * 128;

  v8f acc[8] = {};

  for (int k0 = 0; k0 < D; k0 += 32) {
    // Stage A: 128x32 f32 -> f16, row-major (stride 32 halves)
    {
      const int row = t >> 1, kk = (t & 1) * 16;
      const float* src = X + (size_t)(m0 + row) * D + k0 + kk;
      __builtin_prefetch(src + 32, 0, 0);  // next k-slice -> global_prefetch_b8
#pragma unroll
      for (int j = 0; j < 4; j++) {
        v4f f = *(const v4f*)(src + 4 * j);
        v4h h4;
        h4[0] = (_Float16)f[0]; h4[1] = (_Float16)f[1];
        h4[2] = (_Float16)f[2]; h4[3] = (_Float16)f[3];
        *(v4h*)&sA[row * 32 + kk + 4 * j] = h4;
      }
    }
    // Stage B: [c][k] layout (c = n - n0).  Wcat[k][n] = Wqkv[n/64][k][n%64]
    {
      const int kk = t & 31;
      const int cbase = (t >> 5) * 16;
#pragma unroll
      for (int j = 0; j < 4; j++) {
        const int c = cbase + 4 * j;
        const int n = n0 + c;
        const float* src = W + (size_t)(n >> 6) * (D * HD) + (size_t)(k0 + kk) * HD + (n & 63);
        v4f f = *(const v4f*)src;
#pragma unroll
        for (int i = 0; i < 4; i++) sB[(c + i) * 32 + kk] = (_Float16)f[i];
      }
    }
    __syncthreads();

    const int arow = wave * 16 + lh;
    v16h af = join16(*(const v8h*)&sA[arow * 32 + hi * 16],
                     *(const v8h*)&sA[arow * 32 + hi * 16 + 8]);
#pragma unroll
    for (int g = 0; g < 8; g++) {
      const int c = g * 16 + lh;
      v16h bf = join16(*(const v8h*)&sB[c * 32 + hi * 16],
                       *(const v8h*)&sB[c * 32 + hi * 16 + 8]);
      acc[g] = WMMA_F16(af, bf, acc[g]);
    }
    __syncthreads();
  }

#pragma unroll
  for (int g = 0; g < 8; g++) {
    const int n = n0 + g * 16 + lh;
    const int e = n >> 6, hc = n & 63;
    const float bv = bias[e * 64 + hc];
#pragma unroll
    for (int r = 0; r < 8; r++) {
      const int m = m0 + wave * 16 + r + hi * 8;
      qkv[(size_t)e * M * HD + (size_t)m * HD + hc] = (_Float16)(acc[g][r] + bv);
    }
  }
}

// ---------------------------------------------------------------------------
// Kernel 2: flash attention.  Block = 8 waves; each wave owns a 16-row Q tile.
// K tile (32x64 f16) is double-buffered and staged with ASYNC-to-LDS DMA,
// pipelined one tile ahead; V is register-pipelined one tile ahead and stored
// transposed into LDS.  One barrier per iteration:
//   - sVT[buf] stores happen before the barrier; its readers (same iter) after.
//   - async into sK[buf^1] is issued after the barrier, so the previous
//     iteration's readers of that buffer are already past it.
//   - WAIT_ASYNC(0) before the barrier guarantees sK[buf] (issued last iter)
//     has landed for every wave.
// av ws layout: [b*L + l][h*64 + d] f16 (ready for proj GEMM)
// ---------------------------------------------------------------------------
__global__ __launch_bounds__(256) void flash_attn(const _Float16* __restrict__ qkv,
                                                  _Float16* __restrict__ av) {
  __shared__ __align__(16) _Float16 sK[2][32 * 64];   // [k'][d]
  __shared__ __align__(16) _Float16 sVT[2][64 * 32];  // [d][k']
  __shared__ __align__(16) _Float16 sP[8 * 16 * 32];  // per-wave P scratch

  const int t = threadIdx.x;
  const int wave = t >> 5, lane = t & 31, lh = lane & 15, hi = lane >> 4;
  const int hb = blockIdx.x;         // 0..63
  const int h = hb >> 2, b = hb & 3;
  const int q0 = blockIdx.y * 128 + wave * 16;

  const size_t eStride = (size_t)M * HD;
  const _Float16* Q = qkv + (size_t)h * eStride + (size_t)(b * L) * HD;
  const _Float16* K = qkv + (size_t)(NH + h) * eStride + (size_t)(b * L) * HD;
  const _Float16* V = qkv + (size_t)(2 * NH + h) * eStride + (size_t)(b * L) * HD;

  // Q A-frags for d 0..31 and d 32..63 (held in registers for whole kernel)
  const _Float16* qrow = Q + (size_t)(q0 + lh) * HD;
  const v16h aq0 = join16(*(const v8h*)(qrow + hi * 16), *(const v8h*)(qrow + hi * 16 + 8));
  const v16h aq1 = join16(*(const v8h*)(qrow + 32 + hi * 16), *(const v8h*)(qrow + 32 + hi * 16 + 8));

  v8f acco[4] = {};
  float mi[8], li[8];
#pragma unroll
  for (int r = 0; r < 8; r++) { mi[r] = -1e30f; li[r] = 0.0f; }

  const float sc = 8.0f;  // reference multiplies by sqrt(head_dim)

  const int srow = t >> 3;        // 0..31 : k' row this thread stages
  const int sdd  = (t & 7) * 8;   // 0..56 : d sub-slice (8 halves = 16B)

#if HAS_ASYNC
  v8h vreg;
  // Prologue: tile 0 in flight
  ASYNC_B128(K + (size_t)srow * HD + sdd, &sK[0][srow * 64 + sdd]);
  vreg = *(const v8h*)(V + (size_t)srow * HD + sdd);
#endif

  for (int kt = 0; kt < L; kt += 32) {
    const int buf = (kt >> 5) & 1;

#if HAS_ASYNC
    // Commit register-pipelined V tile (kt) transposed into sVT[buf]
#pragma unroll
    for (int i = 0; i < 8; i++) sVT[buf][(sdd + i) * 32 + srow] = vreg[i];
    WAIT_ASYNC(0);        // K tile kt landed in sK[buf]
    __syncthreads();      // staging visible; prior readers of buf^1 are past
    {
      const int ktn = (kt + 32 < L) ? kt + 32 : kt;  // clamp (tail redundant)
      ASYNC_B128(K + (size_t)(ktn + srow) * HD + sdd, &sK[buf ^ 1][srow * 64 + sdd]);
      vreg = *(const v8h*)(V + (size_t)(ktn + srow) * HD + sdd);
    }
#else
    {
      v8h kv = *(const v8h*)(K + (size_t)(kt + srow) * HD + sdd);
      *(v8h*)&sK[buf][srow * 64 + sdd] = kv;
      v8h vv = *(const v8h*)(V + (size_t)(kt + srow) * HD + sdd);
#pragma unroll
      for (int i = 0; i < 8; i++) sVT[buf][(sdd + i) * 32 + srow] = vv[i];
    }
    __syncthreads();
#endif

    // S (16x32) = Q x K^T : two N-groups of 16, contraction 64 = 2 WMMA each
    v8f s[2];
#pragma unroll
    for (int g = 0; g < 2; g++) {
      const int col = g * 16 + lh;  // k' index = K row
      v16h b0 = join16(*(const v8h*)&sK[buf][col * 64 + hi * 16],
                       *(const v8h*)&sK[buf][col * 64 + hi * 16 + 8]);
      v16h b1 = join16(*(const v8h*)&sK[buf][col * 64 + 32 + hi * 16],
                       *(const v8h*)&sK[buf][col * 64 + 32 + hi * 16 + 8]);
      v8f z = {};
      z = WMMA_F16(aq0, b0, z);
      z = WMMA_F16(aq1, b1, z);
      s[g] = z;
    }

    // Online softmax update (row r lives in lanes 0-15 / 8+r in lanes 16-31)
#pragma unroll
    for (int r = 0; r < 8; r++) {
      float s0 = s[0][r] * sc, s1 = s[1][r] * sc;
      float rm = fmaxf(s0, s1);
      rm = fmaxf(rm, __shfl_xor(rm, 1));
      rm = fmaxf(rm, __shfl_xor(rm, 2));
      rm = fmaxf(rm, __shfl_xor(rm, 4));
      rm = fmaxf(rm, __shfl_xor(rm, 8));
      const float mnew = fmaxf(mi[r], rm);
      const float corr = __expf(mi[r] - mnew);
      const float p0 = __expf(s0 - mnew);
      const float p1 = __expf(s1 - mnew);
      float rs = p0 + p1;
      rs += __shfl_xor(rs, 1);
      rs += __shfl_xor(rs, 2);
      rs += __shfl_xor(rs, 4);
      rs += __shfl_xor(rs, 8);
      li[r] = li[r] * corr + rs;
      mi[r] = mnew;
#pragma unroll
      for (int n = 0; n < 4; n++) acco[n][r] *= corr;
      const int prow = r + hi * 8;
      sP[wave * 512 + prow * 32 + lh]      = (_Float16)p0;
      sP[wave * 512 + prow * 32 + 16 + lh] = (_Float16)p1;
    }

    // Reload P in A-matrix layout; accumulate O += P x V
    {
      const _Float16* pb = &sP[wave * 512];
      v16h ap = join16(*(const v8h*)&pb[lh * 32 + hi * 16],
                       *(const v8h*)&pb[lh * 32 + hi * 16 + 8]);
#pragma unroll
      for (int n = 0; n < 4; n++) {
        const int d = n * 16 + lh;
        v16h bv = join16(*(const v8h*)&sVT[buf][d * 32 + hi * 16],
                         *(const v8h*)&sVT[buf][d * 32 + hi * 16 + 8]);
        acco[n] = WMMA_F16(ap, bv, acco[n]);
      }
    }
#if !HAS_ASYNC
    __syncthreads();
#endif
  }

  // Epilogue: O /= l, store f16 in (b,l, h*64+d) layout
#pragma unroll
  for (int n = 0; n < 4; n++) {
#pragma unroll
    for (int r = 0; r < 8; r++) {
      const int l = q0 + r + hi * 8;
      const float o = acco[n][r] / li[r];
      av[((size_t)(b * L + l)) * (NH * HD) + h * HD + n * 16 + lh] = (_Float16)o;
    }
  }
}

// ---------------------------------------------------------------------------
// Kernel 3: output projection.  out = AV(8192x1024 f16) @ Wp^T + bp, f32 out.
// AV tile staged with async-to-LDS (f16, raw copy); Wp tile converted f32->f16
// manually (contiguous: Wp rows ARE the B-matrix columns).
// ---------------------------------------------------------------------------
__global__ __launch_bounds__(256) void proj_gemm(const _Float16* __restrict__ AV,
                                                 const float* __restrict__ Wp,
                                                 const float* __restrict__ bp,
                                                 float* __restrict__ out) {
  __shared__ __align__(16) _Float16 sA[128 * 32];
  __shared__ __align__(16) _Float16 sB[128 * 32];
  const int t = threadIdx.x;
  const int wave = t >> 5, lane = t & 31, lh = lane & 15, hi = lane >> 4;
  const int m0 = blockIdx.x * 128, n0 = blockIdx.y * 128;

  v8f acc[8] = {};

  for (int k0 = 0; k0 < D; k0 += 32) {
    {
      const int row = t >> 1, kk = (t & 1) * 16;
      const _Float16* src = AV + (size_t)(m0 + row) * D + k0 + kk;
#if HAS_ASYNC
      ASYNC_B128(src,     &sA[row * 32 + kk]);
      ASYNC_B128(src + 8, &sA[row * 32 + kk + 8]);
#else
      *(v8h*)&sA[row * 32 + kk]     = *(const v8h*)src;
      *(v8h*)&sA[row * 32 + kk + 8] = *(const v8h*)(src + 8);
#endif
    }
    {
      const int c = t >> 1, kk = (t & 1) * 16;
      const float* src = Wp + (size_t)(n0 + c) * D + k0 + kk;
#pragma unroll
      for (int j = 0; j < 4; j++) {
        v4f f = *(const v4f*)(src + 4 * j);
        v4h h4;
        h4[0] = (_Float16)f[0]; h4[1] = (_Float16)f[1];
        h4[2] = (_Float16)f[2]; h4[3] = (_Float16)f[3];
        *(v4h*)&sB[c * 32 + kk + 4 * j] = h4;
      }
    }
    WAIT_ASYNC(0);
    __syncthreads();

    const int arow = wave * 16 + lh;
    v16h af = join16(*(const v8h*)&sA[arow * 32 + hi * 16],
                     *(const v8h*)&sA[arow * 32 + hi * 16 + 8]);
#pragma unroll
    for (int g = 0; g < 8; g++) {
      const int c = g * 16 + lh;
      v16h bf = join16(*(const v8h*)&sB[c * 32 + hi * 16],
                       *(const v8h*)&sB[c * 32 + hi * 16 + 8]);
      acc[g] = WMMA_F16(af, bf, acc[g]);
    }
    __syncthreads();
  }

#pragma unroll
  for (int g = 0; g < 8; g++) {
    const int n = n0 + g * 16 + lh;
    const float bv = bp[n];
#pragma unroll
    for (int r = 0; r < 8; r++) {
      const int m = m0 + wave * 16 + r + hi * 8;
      out[(size_t)m * D + n] = acc[g][r] + bv;
    }
  }
}

// ---------------------------------------------------------------------------
extern "C" void kernel_launch(void* const* d_in, const int* in_sizes, int n_in,
                              void* d_out, int out_size, void* d_ws, size_t ws_size,
                              hipStream_t stream) {
  const float* x    = (const float*)d_in[0];  // (4, 2048, 1024)
  const float* Wqkv = (const float*)d_in[1];  // (48, 1024, 64)
  const float* bqkv = (const float*)d_in[2];  // (48, 64)
  const float* Wp   = (const float*)d_in[3];  // (1024, 1024)
  const float* bp   = (const float*)d_in[4];  // (1024,)
  float* out = (float*)d_out;                 // (4, 2048, 1024) f32

  // Workspace: qkv f16 [48][8192][64] then av f16 [8192][1024]
  _Float16* qkv = (_Float16*)d_ws;
  _Float16* av  = qkv + (size_t)48 * M * HD;

  qkv_gemm<<<dim3(M / 128, NQKV / 128), 256, 0, stream>>>(x, Wqkv, bqkv, qkv);
  flash_attn<<<dim3(NH * Bsz, L / 128), 256, 0, stream>>>(qkv, av);
  proj_gemm<<<dim3(M / 128, D / 128), 256, 0, stream>>>(av, Wp, bp, out);
}
